// Model_9620726743391
// MI455X (gfx1250) — matrix operand order repaired and checked by reading, not compiled, not run on hardware
//
#include <hip/hip_runtime.h>
#include <hip/hip_bf16.h>
#include <math.h>

#define BSZ     128
#define SEQ     64
#define DMODEL  512
#define NHEAD   8
#define DH      64
#define NLAYER  2
#define NCOLS   (BSZ * (SEQ + 1))   /* 8320 */
#define MROWS   (BSZ * SEQ)         /* 8192 */
#define NEGINF  (-10000.0f)
#define CHUNK   1024

typedef __attribute__((ext_vector_type(16))) __bf16       v16bf;
typedef __attribute__((ext_vector_type(8)))  float        v8f;
typedef __attribute__((ext_vector_type(4)))  unsigned int u32x4;
typedef __attribute__((ext_vector_type(4)))  float        f32x4;

__device__ __forceinline__ unsigned short f2bf(float f) {
  unsigned int u = __float_as_uint(f);
  u += 0x7fffu + ((u >> 16) & 1u);          // round-to-nearest-even
  return (unsigned short)(u >> 16);
}

// ---------------------------------------------------------------------------
// WMMA GEMM: C[M,N] = A[M,K] * B[N,K]^T   (A,B bf16 row-major lda/ldb = K)
// Block: 256 thr = 8 waves, 128x128 C tile.  Wave (wm 0..3, wn 0..1) owns a
// 32x64 strip = 2x4 grid of v_wmma_f32_16x16x32_bf16 accumulators.
// Register-staged double buffering hides global latency behind the 8 WMMAs.
// ---------------------------------------------------------------------------
__global__ __launch_bounds__(256) void gemm_bf16_kernel(
    const unsigned short* __restrict__ A,   // [gridDim.y*128, K]
    const unsigned short* __restrict__ B,   // [Nn, K]
    float* __restrict__ C,                  // [gridDim.y*128, Nn]
    int Nn, int K)
{
  __shared__ __align__(16) unsigned short As[128][32];   // 8 KB
  __shared__ __align__(16) unsigned short Bs[128][32];   // 8 KB

  const int tid  = threadIdx.x;
  const int lane = tid & 31;
  const int wave = tid >> 5;
  const int wm   = wave >> 1;     // 0..3  -> M offset 32*wm
  const int wn   = wave & 1;      // 0..1  -> N offset 64*wn
  const long blockN = (long)blockIdx.x * 128;
  const long blockM = (long)blockIdx.y * 128;

  v8f acc[2][4];
#pragma unroll
  for (int mi = 0; mi < 2; ++mi)
#pragma unroll
    for (int ni = 0; ni < 4; ++ni) acc[mi][ni] = (v8f){};

  // loader mapping: 2 threads per tile row, 32B (two b128) each per matrix
  const int ldrow = tid >> 1;          // 0..127
  const int ldcol = (tid & 1) * 16;    // 0 or 16 (ushort units)
  const unsigned short* Arow = A + (blockM + ldrow) * (long)K + ldcol;
  const unsigned short* Brow = B + (blockN + ldrow) * (long)K + ldcol;

  u32x4 ra0, ra1, rb0, rb1;
  ra0 = *(const u32x4*)(Arow + 0);
  ra1 = *(const u32x4*)(Arow + 8);
  rb0 = *(const u32x4*)(Brow + 0);
  rb1 = *(const u32x4*)(Brow + 8);

  for (int k0 = 0; k0 < K; k0 += 32) {
    *(u32x4*)&As[ldrow][ldcol]     = ra0;
    *(u32x4*)&As[ldrow][ldcol + 8] = ra1;
    *(u32x4*)&Bs[ldrow][ldcol]     = rb0;
    *(u32x4*)&Bs[ldrow][ldcol + 8] = rb1;
    __syncthreads();

    // stage next k-tile into registers while WMMAs run on the current one
    if (k0 + 32 < K) {
      ra0 = *(const u32x4*)(Arow + k0 + 32);
      ra1 = *(const u32x4*)(Arow + k0 + 40);
      rb0 = *(const u32x4*)(Brow + k0 + 32);
      rb1 = *(const u32x4*)(Brow + k0 + 40);
      if (k0 + 64 < K) {                       // L2 prefetch two tiles ahead
        __builtin_prefetch(Arow + k0 + 64, 0, 1);
        __builtin_prefetch(Brow + k0 + 64, 0, 1);
      }
    }

    // A fragments (ISA 7.12.2 bf16 A layout: lanes 0-15 K=0..7/16..23,
    // lanes 16-31 K=8..15/24..31)
    const int akh = (lane >> 4) * 8;
    union { u32x4 u[2]; v16bf v; } fa[2];
#pragma unroll
    for (int mi = 0; mi < 2; ++mi) {
      const int am = wm * 32 + mi * 16 + (lane & 15);
      fa[mi].u[0] = *(const u32x4*)&As[am][akh];
      fa[mi].u[1] = *(const u32x4*)&As[am][akh + 16];
    }
    // B fragments: Bs is N-major so each lane's column has contiguous K
    const int bkh = (lane >> 4) * 8;
    union { u32x4 u[2]; v16bf v; } fb[4];
#pragma unroll
    for (int ni = 0; ni < 4; ++ni) {
      const int bn = wn * 64 + ni * 16 + (lane & 15);
      fb[ni].u[0] = *(const u32x4*)&Bs[bn][bkh];
      fb[ni].u[1] = *(const u32x4*)&Bs[bn][bkh + 16];
    }

#pragma unroll
    for (int mi = 0; mi < 2; ++mi)
#pragma unroll
      for (int ni = 0; ni < 4; ++ni)
        acc[mi][ni] = __builtin_amdgcn_wmma_f32_16x16x32_bf16(
            false, fa[mi].v, false, fb[ni].v, (short)0, acc[mi][ni],
            false, false);
    __syncthreads();
  }

  // C/D layout: VGPR r -> M = r + (lane>=16 ? 8 : 0); N = lane & 15
#pragma unroll
  for (int mi = 0; mi < 2; ++mi) {
    const long mbase = blockM + wm * 32 + mi * 16 + ((lane >> 4) * 8);
#pragma unroll
    for (int ni = 0; ni < 4; ++ni) {
      const long nbase = blockN + wn * 64 + ni * 16 + (lane & 15);
#pragma unroll
      for (int r = 0; r < 8; ++r)
        C[(mbase + r) * Nn + nbase] = acc[mi][ni][r];
    }
  }
}

// ---------------------------------------------------------------------------
// Weight transpose + fp32->bf16:  in[K,Nn] -> outT[Nn,K]
// ---------------------------------------------------------------------------
__global__ void wt_kernel(const float* __restrict__ in,
                          unsigned short* __restrict__ outT, int K, int Nn)
{
  long i = (long)blockIdx.x * blockDim.x + threadIdx.x;
  if (i >= (long)K * Nn) return;
  int n = (int)(i / K);
  int k = (int)(i % K);
  outT[i] = f2bf(in[(long)k * Nn + n]);
}

// ---------------------------------------------------------------------------
// Embedding gather (+pos add for encoder input)
// ---------------------------------------------------------------------------
__global__ void gather_kernel(const int* __restrict__ items,
                              const float* __restrict__ emb_table,
                              const float* __restrict__ pos_emb,
                              float* __restrict__ s_f32,
                              unsigned short* __restrict__ s_bf,
                              float* __restrict__ x_in)
{
  long i = (long)blockIdx.x * blockDim.x + threadIdx.x;
  if (i >= (long)NCOLS * DMODEL) return;
  int rowi = (int)(i >> 9);
  int c    = (int)(i & 511);
  int item = items[rowi];
  float v = emb_table[(long)item * DMODEL + c];
  s_f32[i] = v;
  s_bf[i]  = f2bf(v);
  int bb = rowi / (SEQ + 1), j = rowi % (SEQ + 1);
  if (j < SEQ)
    x_in[((long)bb * SEQ + j) * DMODEL + c] = v + pos_emb[(long)j * DMODEL + c];
}

__global__ void debias_kernel(const int* __restrict__ items,
                              const float* __restrict__ pop,
                              float* __restrict__ debias)
{
  int i = blockIdx.x * blockDim.x + threadIdx.x;
  if (i >= NCOLS) return;
  debias[i] = __logf(pop[items[i]]);
}

__global__ void member_kernel(const int* __restrict__ items,
                              unsigned char* __restrict__ member)
{
  long i = (long)blockIdx.x * blockDim.x + threadIdx.x;
  if (i >= (long)BSZ * NCOLS) return;
  int bb  = (int)(i / NCOLS);
  int col = (int)(i % NCOLS);
  int item = items[col];
  const int* lst = items + bb * (SEQ + 1);
  unsigned char m = 0;
  for (int t = 0; t < SEQ + 1; ++t) m |= (unsigned char)(lst[t] == item);
  member[i] = m;
}

__global__ void colpad_kernel(const float* __restrict__ log_mask,
                              unsigned char* __restrict__ colpad)
{
  int col = blockIdx.x * blockDim.x + threadIdx.x;
  if (col >= NCOLS) return;
  int bb = col / (SEQ + 1), j = col % (SEQ + 1);
  colpad[col] = (unsigned char)((j < SEQ) && (log_mask[bb * SEQ + j] == 0.0f));
}

// ---------------------------------------------------------------------------
// x = LayerNorm(a [+ b]);  writes f32 and bf16 copies.  One row per block.
// ---------------------------------------------------------------------------
__global__ __launch_bounds__(128) void add_ln_kernel(
    const float* __restrict__ a, const float* __restrict__ badd,
    float* __restrict__ out_f, unsigned short* __restrict__ out_bf)
{
  __shared__ float red[128];
  const long r  = blockIdx.x;
  const int tid = threadIdx.x;
  f32x4 v = *(const f32x4*)(a + r * DMODEL + tid * 4);
  if (badd) {
    f32x4 w = *(const f32x4*)(badd + r * DMODEL + tid * 4);
    v += w;
  }
  red[tid] = v[0] + v[1] + v[2] + v[3];
  __syncthreads();
  for (int st = 64; st > 0; st >>= 1) {
    if (tid < st) red[tid] += red[tid + st];
    __syncthreads();
  }
  float mean = red[0] * (1.0f / DMODEL);
  __syncthreads();
  f32x4 d = v - mean;
  red[tid] = d[0]*d[0] + d[1]*d[1] + d[2]*d[2] + d[3]*d[3];
  __syncthreads();
  for (int st = 64; st > 0; st >>= 1) {
    if (tid < st) red[tid] += red[tid + st];
    __syncthreads();
  }
  float rstd = rsqrtf(red[0] * (1.0f / DMODEL) + 1e-5f);
  for (int j = 0; j < 4; ++j) {
    float o = d[j] * rstd;
    out_f[r * DMODEL + tid * 4 + j]  = o;
    out_bf[r * DMODEL + tid * 4 + j] = f2bf(o);
  }
}

// ---------------------------------------------------------------------------
// Attention: one (batch, head) per block.  S=64, dh=64; causal + pad mask.
// ---------------------------------------------------------------------------
__global__ __launch_bounds__(256) void attn_kernel(
    const float* __restrict__ qkv,        // [MROWS, 3*DMODEL]
    const float* __restrict__ log_mask,   // [BSZ, SEQ]
    unsigned short* __restrict__ o_bf)    // [MROWS, DMODEL]
{
  __shared__ float Qs[SEQ][DH];
  __shared__ float Ks[SEQ][DH];
  __shared__ float Vs[SEQ][DH];
  __shared__ float Ss[SEQ][SEQ];
  const int b   = blockIdx.x >> 3;
  const int h   = blockIdx.x & 7;
  const int tid = threadIdx.x;
  const int row = tid >> 2;          // 0..63
  const int seg = (tid & 3) * 16;    // 0,16,32,48

  const float* base = qkv + (long)(b * SEQ + row) * (3 * DMODEL);
  for (int i = 0; i < 16; ++i) {
    Qs[row][seg + i] = base[h * DH + seg + i];
    Ks[row][seg + i] = base[DMODEL + h * DH + seg + i];
    Vs[row][seg + i] = base[2 * DMODEL + h * DH + seg + i];
  }
  __syncthreads();

  for (int kk = seg; kk < seg + 16; ++kk) {
    float dot = 0.f;
    for (int d = 0; d < DH; ++d) dot += Qs[row][d] * Ks[kk][d];
    bool ok = (kk <= row) && (log_mask[b * SEQ + kk] != 0.0f);
    Ss[row][kk] = dot * 0.125f + (ok ? 0.0f : NEGINF);   // 1/sqrt(64)
  }
  __syncthreads();

  if (tid < SEQ) {
    float mx = Ss[tid][0];
    for (int kk = 1; kk < SEQ; ++kk) mx = fmaxf(mx, Ss[tid][kk]);
    float sum = 0.f;
    for (int kk = 0; kk < SEQ; ++kk) {
      float e = __expf(Ss[tid][kk] - mx);
      Ss[tid][kk] = e;
      sum += e;
    }
    float inv = 1.0f / sum;
    for (int kk = 0; kk < SEQ; ++kk) Ss[tid][kk] *= inv;
  }
  __syncthreads();

  for (int d = seg; d < seg + 16; ++d) {
    float acc = 0.f;
    for (int kk = 0; kk < SEQ; ++kk) acc += Ss[row][kk] * Vs[kk][d];
    o_bf[(long)(b * SEQ + row) * DMODEL + h * DH + d] = f2bf(acc);
  }
}

__global__ void gelu_kernel(const float* __restrict__ in,
                            unsigned short* __restrict__ out_bf, long n)
{
  long i = (long)blockIdx.x * blockDim.x + threadIdx.x;
  if (i >= n) return;
  float x = in[i];
  float c = 0.7978845608028654f * (x + 0.044715f * x * x * x);
  out_bf[i] = f2bf(0.5f * x * (1.0f + tanhf(c)));
}

// ---------------------------------------------------------------------------
// Fused masked log-softmax NLL over one chunk of logit rows.
// ---------------------------------------------------------------------------
__global__ __launch_bounds__(256) void loss_kernel(
    const float* __restrict__ logits,     // [CHUNK, NCOLS]
    const float* __restrict__ log_mask,
    const float* __restrict__ debias,
    const unsigned char* __restrict__ member,   // [BSZ, NCOLS]
    const unsigned char* __restrict__ colpad,   // [NCOLS]
    int r0, float* __restrict__ accum)
{
  const int rr = blockIdx.x;
  const int r  = r0 + rr;
  const int b  = r / SEQ, s = r % SEQ;
  if (log_mask[b * SEQ + s] == 0.0f) return;   // invalid row contributes 0
  const int target = b * (SEQ + 1) + s + 1;
  const float* Lg = logits + (long)rr * NCOLS;
  const unsigned char* mem = member + (long)b * NCOLS;
  const int tid = threadIdx.x;

  __shared__ float red[256];
  __shared__ float s_t;
  float mx = -3.0e38f;
  for (int col = tid; col < NCOLS; col += 256) {
    bool masked = colpad[col] || (mem[col] && col != target);
    float v = masked ? NEGINF : (Lg[col] - debias[col]);
    if (col == target) s_t = v;
    mx = fmaxf(mx, v);
  }
  red[tid] = mx;
  __syncthreads();
  for (int st = 128; st > 0; st >>= 1) {
    if (tid < st) red[tid] = fmaxf(red[tid], red[tid + st]);
    __syncthreads();
  }
  mx = red[0];
  __syncthreads();

  float sum = 0.f;
  for (int col = tid; col < NCOLS; col += 256) {
    bool masked = colpad[col] || (mem[col] && col != target);
    float v = masked ? NEGINF : (Lg[col] - debias[col]);
    sum += __expf(v - mx);
  }
  red[tid] = sum;
  __syncthreads();
  for (int st = 128; st > 0; st >>= 1) {
    if (tid < st) red[tid] += red[tid + st];
    __syncthreads();
  }
  if (tid == 0) {
    float nll = -(s_t - mx - __logf(red[0]));
    atomicAdd(accum, nll);
  }
}

__global__ void init_kernel(float* accum) {
  if (threadIdx.x < 2) accum[threadIdx.x] = 0.0f;
}
__global__ void valid_kernel(const float* __restrict__ log_mask, float* accum) {
  int i = blockIdx.x * blockDim.x + threadIdx.x;
  if (i < MROWS && log_mask[i] != 0.0f) atomicAdd(accum + 1, 1.0f);
}
__global__ void final_kernel(const float* accum, float* out) {
  if (threadIdx.x == 0) out[0] = accum[0] / accum[1];
}

// ---------------------------------------------------------------------------
extern "C" void kernel_launch(void* const* d_in, const int* in_sizes, int n_in,
                              void* d_out, int out_size, void* d_ws, size_t ws_size,
                              hipStream_t stream)
{
  (void)in_sizes; (void)n_in; (void)out_size; (void)ws_size;
  const int*   items     = (const int*)d_in[0];
  const float* log_mask  = (const float*)d_in[1];
  const float* emb_table = (const float*)d_in[2];
  const float* pos_emb   = (const float*)d_in[3];
  const float* qkv_w     = (const float*)d_in[4];
  const float* out_w     = (const float*)d_in[5];
  const float* ffn_w1    = (const float*)d_in[6];
  const float* ffn_w2    = (const float*)d_in[7];
  const float* pop_prob  = (const float*)d_in[8];
  float* out = (float*)d_out;

  // ---- carve workspace (all offsets 256B aligned) ----
  char* p = (char*)d_ws;
  auto carve = [&](size_t bytes) -> void* {
    void* q = (void*)p;
    p += (bytes + 255) & ~(size_t)255;
    return q;
  };
  float*          s_f32   = (float*)         carve((size_t)NCOLS * DMODEL * 4);
  unsigned short* s_bf    = (unsigned short*)carve((size_t)NCOLS * DMODEL * 2);
  float*          debias  = (float*)         carve((size_t)NCOLS * 4);
  float*          x_in    = (float*)         carve((size_t)MROWS * DMODEL * 4);
  float*          x       = (float*)         carve((size_t)MROWS * DMODEL * 4);
  unsigned short* x_bf    = (unsigned short*)carve((size_t)MROWS * DMODEL * 2);
  float*          qkv     = (float*)         carve((size_t)MROWS * 3 * DMODEL * 4);
  unsigned short* o_bf    = (unsigned short*)carve((size_t)MROWS * DMODEL * 2);
  float*          t2      = (float*)         carve((size_t)MROWS * DMODEL * 4);
  float*          t1      = (float*)         carve((size_t)MROWS * 4 * DMODEL * 4);
  unsigned short* t1_bf   = (unsigned short*)carve((size_t)MROWS * 4 * DMODEL * 2);
  unsigned short* wT_qkv  = (unsigned short*)carve((size_t)NLAYER * 3 * DMODEL * DMODEL * 2);
  unsigned short* wT_out  = (unsigned short*)carve((size_t)NLAYER * DMODEL * DMODEL * 2);
  unsigned short* wT_f1   = (unsigned short*)carve((size_t)NLAYER * 4 * DMODEL * DMODEL * 2);
  unsigned short* wT_f2   = (unsigned short*)carve((size_t)NLAYER * DMODEL * 4 * DMODEL * 2);
  unsigned char*  member  = (unsigned char*) carve((size_t)BSZ * NCOLS);
  unsigned char*  colpad  = (unsigned char*) carve((size_t)NCOLS);
  float*          logitsC = (float*)         carve((size_t)CHUNK * NCOLS * 4);
  float*          accum   = (float*)         carve(256);

  // ---- weights: transpose + bf16 once per launch ----
  for (int l = 0; l < NLAYER; ++l) {
    long e;
    e = (long)DMODEL * 3 * DMODEL;
    wt_kernel<<<(unsigned)((e + 255) / 256), 256, 0, stream>>>(qkv_w + l * e, wT_qkv + l * e, DMODEL, 3 * DMODEL);
    e = (long)DMODEL * DMODEL;
    wt_kernel<<<(unsigned)((e + 255) / 256), 256, 0, stream>>>(out_w + l * e, wT_out + l * e, DMODEL, DMODEL);
    e = (long)DMODEL * 4 * DMODEL;
    wt_kernel<<<(unsigned)((e + 255) / 256), 256, 0, stream>>>(ffn_w1 + l * e, wT_f1 + l * e, DMODEL, 4 * DMODEL);
    wt_kernel<<<(unsigned)((e + 255) / 256), 256, 0, stream>>>(ffn_w2 + l * e, wT_f2 + l * e, 4 * DMODEL, DMODEL);
  }

  gather_kernel<<<((long)NCOLS * DMODEL + 255) / 256, 256, 0, stream>>>(
      items, emb_table, pos_emb, s_f32, s_bf, x_in);
  debias_kernel<<<(NCOLS + 255) / 256, 256, 0, stream>>>(items, pop_prob, debias);
  member_kernel<<<(unsigned)(((long)BSZ * NCOLS + 255) / 256), 256, 0, stream>>>(items, member);
  colpad_kernel<<<(NCOLS + 255) / 256, 256, 0, stream>>>(log_mask, colpad);
  init_kernel<<<1, 32, 0, stream>>>(accum);
  valid_kernel<<<(MROWS + 255) / 256, 256, 0, stream>>>(log_mask, accum);

  // x = LN(embs + pos)
  add_ln_kernel<<<MROWS, 128, 0, stream>>>(x_in, nullptr, x, x_bf);

  for (int l = 0; l < NLAYER; ++l) {
    // qkv = x @ qkv_w[l]
    gemm_bf16_kernel<<<dim3(3 * DMODEL / 128, MROWS / 128), 256, 0, stream>>>(
        x_bf, wT_qkv + (long)l * 3 * DMODEL * DMODEL, qkv, 3 * DMODEL, DMODEL);
    // multi-head causal attention
    attn_kernel<<<BSZ * NHEAD, 256, 0, stream>>>(qkv, log_mask, o_bf);
    // t2 = o @ out_w[l];  x = LN(x + t2)
    gemm_bf16_kernel<<<dim3(DMODEL / 128, MROWS / 128), 256, 0, stream>>>(
        o_bf, wT_out + (long)l * DMODEL * DMODEL, t2, DMODEL, DMODEL);
    add_ln_kernel<<<MROWS, 128, 0, stream>>>(x, t2, x, x_bf);
    // t1 = gelu(x @ ffn_w1[l])
    gemm_bf16_kernel<<<dim3(4 * DMODEL / 128, MROWS / 128), 256, 0, stream>>>(
        x_bf, wT_f1 + (long)l * 4 * DMODEL * DMODEL, t1, 4 * DMODEL, DMODEL);
    gelu_kernel<<<(unsigned)(((long)MROWS * 4 * DMODEL + 255) / 256), 256, 0, stream>>>(
        t1, t1_bf, (long)MROWS * 4 * DMODEL);
    // t2 = t1 @ ffn_w2[l];  x = LN(x + t2)
    gemm_bf16_kernel<<<dim3(DMODEL / 128, MROWS / 128), 256, 0, stream>>>(
        t1_bf, wT_f2 + (long)l * DMODEL * 4 * DMODEL, t2, DMODEL, 4 * DMODEL);
    add_ln_kernel<<<MROWS, 128, 0, stream>>>(x, t2, x, x_bf);
  }

  // logits = prec @ score_embs^T, streamed in row chunks + fused loss
  for (int r0 = 0; r0 < MROWS; r0 += CHUNK) {
    gemm_bf16_kernel<<<dim3(NCOLS / 128, CHUNK / 128), 256, 0, stream>>>(
        x_bf + (long)r0 * DMODEL, s_bf, logitsC, NCOLS, DMODEL);
    loss_kernel<<<CHUNK, 256, 0, stream>>>(logitsC, log_mask, debias, member, colpad, r0, accum);
  }
  final_kernel<<<1, 32, 0, stream>>>(accum, out);
}
